// PointPillarModel_34248069218719
// MI455X (gfx1250) — compile-verified, matrix operand order checked by reading, not run on hardware
//
#include <hip/hip_runtime.h>
#include <hip/hip_bf16.h>

// ---------------------------------------------------------------------------
// PointPillars on MI455X (gfx1250, wave32, WMMA)
//   PFN -> scatter -> 3x conv3x3(+relu) implicit-GEMM via v_wmma_f32_16x16x32_f16
//   -> 1x1 heads (cls 10ch, reg 7ch) written f32 to d_out.
// Intermediates f16 NHWC in workspace; conv weights pre-packed into WMMA
// B-fragment order (one contiguous 32B load per lane). Halo tile staged in
// LDS via GLOBAL_LOAD_ASYNC_TO_LDS_B128 (ASYNCcnt) when the builtin exists.
// Each wave owns BOTH 16-pixel M-tiles: every B fragment feeds two WMMAs.
// ---------------------------------------------------------------------------

typedef __attribute__((ext_vector_type(16))) _Float16 v16h;
typedef __attribute__((ext_vector_type(8)))  _Float16 v8h;
typedef __attribute__((ext_vector_type(8)))  float    v8f;
typedef __attribute__((ext_vector_type(4)))  int      v4i;

#define HH 512
#define WW 512
#define PP 20000

#define AS1 __attribute__((address_space(1)))
#define AS3 __attribute__((address_space(3)))

#if defined(__has_builtin)
#  if __has_builtin(__builtin_amdgcn_global_load_async_to_lds_b128) && \
      __has_builtin(__builtin_amdgcn_s_wait_asynccnt)
#    define USE_ASYNC_LDS 1
#  endif
#endif
#ifndef USE_ASYNC_LDS
#  define USE_ASYNC_LDS 0
#endif

// ---------------------------------------------------------------- zero f16 --
__global__ void zero_f16_kernel(_Float16* __restrict__ ptr, long n8) {
    long i = (long)blockIdx.x * blockDim.x + threadIdx.x;
    long stride = (long)gridDim.x * blockDim.x;
    for (; i < n8; i += stride) {
        *(int4*)(ptr + i * 8) = make_int4(0, 0, 0, 0);
    }
}

// ------------------------------------------------------- weight repacking ---
// OIHW f32  ->  [tap=ky*3+kx][kc=Cin/32][cout][32 k] f16  (B-fragment order)
__global__ void pack_weights_kernel(const float* __restrict__ w,
                                    _Float16* __restrict__ wpack,
                                    int cin, int cout) {
    int total = cout * cin * 9;
    int kchunks = cin >> 5;
    for (int idx = blockIdx.x * blockDim.x + threadIdx.x; idx < total;
         idx += gridDim.x * blockDim.x) {
        int kx = idx % 3;
        int ky = (idx / 3) % 3;
        int i  = (idx / 9) % cin;
        int o  = idx / (9 * cin);
        int t  = ky * 3 + kx;
        int kc = i >> 5;
        int ki = i & 31;
        long po = (((long)(t * kchunks + kc) * cout + o) << 5) + ki;
        wpack[po] = (_Float16)w[idx];
    }
}

// --------------------------------------------- PFN + scatter (tiny layer) ---
// one block per pillar, 64 threads = 64 output channels
__global__ __launch_bounds__(64)
void pfn_scatter_kernel(const float* __restrict__ voxels,   // [P][32][4]
                        const int*   __restrict__ coords,   // [P][2] (x,y)
                        const float* __restrict__ lin_w,    // [4][64]
                        const float* __restrict__ gamma,
                        const float* __restrict__ beta,
                        const float* __restrict__ mean,
                        const float* __restrict__ var,
                        _Float16* __restrict__ canvas)      // [H][W][64] f16
{
    __shared__ float vx[32 * 4];
    const int p = blockIdx.x;
    const int c = threadIdx.x;
    for (int i = c; i < 128; i += 64) vx[i] = voxels[(long)p * 128 + i];
    __syncthreads();

    const float sc = gamma[c] * rsqrtf(var[c] + 1e-5f);
    const float sh = beta[c] - mean[c] * sc;
    const float w0 = lin_w[0 * 64 + c];
    const float w1 = lin_w[1 * 64 + c];
    const float w2 = lin_w[2 * 64 + c];
    const float w3 = lin_w[3 * 64 + c];

    float mx = 0.0f;   // relu*mask outputs are >= 0
#pragma unroll
    for (int n = 0; n < 32; ++n) {
        float a0 = vx[n * 4 + 0], a1 = vx[n * 4 + 1];
        float a2 = vx[n * 4 + 2], a3 = vx[n * 4 + 3];
        float x = a0 * w0 + a1 * w1 + a2 * w2 + a3 * w3;
        x = x * sc + sh;
        x = x > 0.0f ? x : 0.0f;
        if (a0 == 0.0f) x = 0.0f;          // mask = (voxels[...,0] != 0)
        mx = x > mx ? x : mx;
    }
    const int cx = coords[p * 2 + 0];
    const int cy = coords[p * 2 + 1];
    canvas[((long)cy * WW + cx) * 64 + c] = (_Float16)mx;
}

// ------------------------------------------ conv3x3 + bias + relu (WMMA) ----
// NHWC f16 in/out. Block = 256 threads (8 wave32 waves).
// Block covers 32 consecutive x-pixels of one row, all Cout channels.
// Wave w owns N-tiles w, w+8, ...; each wave computes BOTH 16-pixel M-tiles
// with two accumulators so every B fragment is reused by two WMMAs.
template <int CIN_, int COUT_>
__global__ __launch_bounds__(256)
void conv3x3_relu_wmma(const _Float16* __restrict__ in,     // [H][W][CIN_]
                       const _Float16* __restrict__ wpack,  // [9][CIN_/32][COUT_][32]
                       const float*    __restrict__ bias,   // [COUT_]
                       _Float16*       __restrict__ out)    // [H][W][COUT_]
{
    constexpr int KC = CIN_ / 32;
    __shared__ _Float16 smem[3 * 34 * CIN_];   // halo tile (y-1..y+1, x-1..x+32)

    const int tid   = threadIdx.x;
    const int lane  = tid & 31;
    const int wave  = tid >> 5;
    const int tileX = blockIdx.x * 32;
    const int y     = blockIdx.y;

    // cooperative halo fill, 16B (8 x f16) chunks, zero-padded at borders.
    // Interior chunks go global->LDS directly on the async path (ASYNCcnt).
    constexpr int C8     = CIN_ / 8;
    constexpr int NCHUNK = 3 * 34 * C8;
    for (int idx = tid; idx < NCHUNK; idx += 256) {
        int c8 = idx % C8;
        int cx = (idx / C8) % 34;
        int r  = idx / (C8 * 34);
        int gx = tileX + cx - 1;
        int gy = y + r - 1;
        _Float16* lp = smem + ((long)r * 34 + cx) * CIN_ + c8 * 8;
        if (gx >= 0 && gx < WW && gy >= 0 && gy < HH) {
            const _Float16* gp = in + ((long)gy * WW + gx) * CIN_ + c8 * 8;
#if USE_ASYNC_LDS
            __builtin_amdgcn_global_load_async_to_lds_b128(
                (AS1 v4i*)gp, (AS3 v4i*)lp, 0, 0);
#else
            *(int4*)lp = *(const int4*)gp;
#endif
        } else {
            *(int4*)lp = make_int4(0, 0, 0, 0);
        }
    }
#if USE_ASYNC_LDS
    __builtin_amdgcn_s_wait_asynccnt(0);
#endif
    __syncthreads();

    const int m     = lane & 15;
    const int khalf = lane >> 4;

    for (int nt = wave; nt < COUT_ / 16; nt += 8) {
        const int nbase = nt * 16;
        v8f acc0 = {};
        v8f acc1 = {};
        // hint the next N-tile's weight block toward the caches early
        __builtin_prefetch(wpack + (((long)nbase + 128 + m) << 5), 0, 1);

#pragma unroll
        for (int dy = 0; dy < 3; ++dy) {
#pragma unroll
            for (int dx = 0; dx < 3; ++dx) {
                const int t = dy * 3 + dx;
#pragma unroll
                for (int kc = 0; kc < KC; ++kc) {
                    const _Float16* bp = wpack +
                        (((long)(t * KC + kc) * COUT_ + nbase + m) << 5) + khalf * 16;
                    v16h b = *(const v16h*)bp;

                    const _Float16* ap0 = smem +
                        ((long)dy * 34 + (m + dx)) * CIN_ + kc * 32 + khalf * 16;
                    const _Float16* ap1 = ap0 + (long)16 * CIN_;   // second M tile
                    v16h a0 = *(const v16h*)ap0;
                    v16h a1 = *(const v16h*)ap1;

                    acc0 = __builtin_amdgcn_wmma_f32_16x16x32_f16(
                        false, a0, false, b, (short)0, acc0, false, false);
                    acc1 = __builtin_amdgcn_wmma_f32_16x16x32_f16(
                        false, a1, false, b, (short)0, acc1, false, false);
                }
            }
        }

        // D layout: n = lane&15, m = v + 8*(lane>>4)
        const int   n  = nbase + m;
        const float bv = bias[n];
        const int   mb = khalf * 8;
#pragma unroll
        for (int v = 0; v < 8; ++v) {
            float r0 = acc0[v] + bv;
            float r1 = acc1[v] + bv;
            r0 = r0 > 0.0f ? r0 : 0.0f;
            r1 = r1 > 0.0f ? r1 : 0.0f;
            int gx0 = tileX + mb + v;
            out[((long)y * WW + gx0) * COUT_ + n]        = (_Float16)r0;
            out[((long)y * WW + gx0 + 16) * COUT_ + n]   = (_Float16)r1;
        }
    }
}

// ------------------------------------------------------- 1x1 conv heads -----
// h3 [H][W][256] f16 -> d_out: cls [10][H][W] f32 then reg [7][H][W] f32
__global__ __launch_bounds__(256)
void heads_1x1_kernel(const _Float16* __restrict__ h3,
                      const float* __restrict__ cls_w,   // [10][256]
                      const float* __restrict__ cls_b,   // [10]
                      const float* __restrict__ reg_w,   // [7][256]
                      const float* __restrict__ reg_b,   // [7]
                      float* __restrict__ outp)
{
    __shared__ float wl[17 * 256];
    for (int i = threadIdx.x; i < 17 * 256; i += 256)
        wl[i] = (i < 10 * 256) ? cls_w[i] : reg_w[i - 10 * 256];
    __syncthreads();

    const int p = blockIdx.x * 256 + threadIdx.x;       // pixel id
    const _Float16* row = h3 + (long)p * 256;

    float acc[17];
#pragma unroll
    for (int j = 0; j < 17; ++j) acc[j] = 0.0f;

    for (int c = 0; c < 256; c += 8) {
        v8h hv = *(const v8h*)(row + c);
#pragma unroll
        for (int u = 0; u < 8; ++u) {
            float v = (float)hv[u];
#pragma unroll
            for (int j = 0; j < 17; ++j) acc[j] += v * wl[j * 256 + c + u];
        }
    }

    const int HW_ = HH * WW;
#pragma unroll
    for (int j = 0; j < 10; ++j) outp[(long)j * HW_ + p] = acc[j] + cls_b[j];
#pragma unroll
    for (int j = 0; j < 7; ++j)
        outp[(long)(10 + j) * HW_ + p] = acc[10 + j] + reg_b[j];
}

// ---------------------------------------------------------------------------
extern "C" void kernel_launch(void* const* d_in, const int* in_sizes, int n_in,
                              void* d_out, int out_size, void* d_ws, size_t ws_size,
                              hipStream_t stream) {
    const float* voxels  = (const float*)d_in[0];
    const int*   coords  = (const int*)d_in[1];
    const float* lin_w   = (const float*)d_in[2];
    const float* bn_g    = (const float*)d_in[3];
    const float* bn_b    = (const float*)d_in[4];
    const float* bn_m    = (const float*)d_in[5];
    const float* bn_v    = (const float*)d_in[6];
    const float* conv1_w = (const float*)d_in[7];
    const float* conv1_b = (const float*)d_in[8];
    const float* conv2_w = (const float*)d_in[9];
    const float* conv2_b = (const float*)d_in[10];
    const float* conv3_w = (const float*)d_in[11];
    const float* conv3_b = (const float*)d_in[12];
    const float* cls_w   = (const float*)d_in[13];
    const float* cls_b   = (const float*)d_in[14];
    const float* reg_w   = (const float*)d_in[15];
    const float* reg_b   = (const float*)d_in[16];

    char* ws = (char*)d_ws;
    // f16 workspace layout
    _Float16* canvas = (_Float16*)(ws + 0);                       //  64ch: 32 MB
    _Float16* h1     = (_Float16*)(ws + 33554432L);               //  64ch: 32 MB
    _Float16* h2     = (_Float16*)(ws + 67108864L);               // 128ch: 64 MB
    _Float16* h3     = (_Float16*)(ws + 134217728L);              // 256ch: 128 MB
    _Float16* wp1    = (_Float16*)(ws + 268435456L);              // 72 KB
    _Float16* wp2    = (_Float16*)(ws + 268509184L);              // 144 KB
    _Float16* wp3    = (_Float16*)(ws + 268656640L);              // 576 KB

    // 1) zero the scatter canvas (512*512*64 f16 = 2,097,152 x 16B chunks)
    zero_f16_kernel<<<4096, 256, 0, stream>>>(canvas, (long)HH * WW * 64 / 8);

    // 2) repack conv weights into WMMA B-fragment order (f16)
    pack_weights_kernel<<<256, 256, 0, stream>>>(conv1_w, wp1, 64, 64);
    pack_weights_kernel<<<256, 256, 0, stream>>>(conv2_w, wp2, 64, 128);
    pack_weights_kernel<<<512, 256, 0, stream>>>(conv3_w, wp3, 128, 256);

    // 3) PFN + scatter
    pfn_scatter_kernel<<<PP, 64, 0, stream>>>(voxels, coords, lin_w,
                                              bn_g, bn_b, bn_m, bn_v, canvas);

    // 4) backbone: 3x conv3x3 + relu, implicit GEMM via WMMA
    dim3 cgrid(WW / 32, HH, 1);
    conv3x3_relu_wmma<64, 64 ><<<cgrid, 256, 0, stream>>>(canvas, wp1, conv1_b, h1);
    conv3x3_relu_wmma<64, 128><<<cgrid, 256, 0, stream>>>(h1,     wp2, conv2_b, h2);
    conv3x3_relu_wmma<128,256><<<cgrid, 256, 0, stream>>>(h2,     wp3, conv3_b, h3);

    // 5) heads (1x1 convs), f32 output into d_out (cls then reg, NCHW flat)
    heads_1x1_kernel<<<(HH * WW) / 256, 256, 0, stream>>>(
        h3, cls_w, cls_b, reg_w, reg_b, (float*)d_out);
}
